// LSTMBlock_14499809591329
// MI455X (gfx1250) — matrix operand order repaired and checked by reading, not compiled, not run on hardware
//
#include <hip/hip_runtime.h>
#include <hip/hip_bf16.h>

// ---------------------------------------------------------------------------
// LSTM block for MI455X (gfx1250, wave32, WMMA).
// GEMMs: v_wmma_f32_16x16x32_bf16 with LDS double-buffered tiles staged via
// GLOBAL_LOAD_ASYNC_TO_LDS_B128 (ASYNCcnt) when the builtin is available.
// LSTM recurrent W_hh tiles are also staged to LDS asynchronously per step.
// ---------------------------------------------------------------------------

typedef __attribute__((ext_vector_type(16))) __bf16 v16bf;
typedef __attribute__((ext_vector_type(8)))  __bf16 v8bf;
typedef __attribute__((ext_vector_type(4)))  __bf16 v4bf;
typedef __attribute__((ext_vector_type(8)))  float  v8f;
typedef int v4i __attribute__((vector_size(16)));

constexpr int Bn = 8;
constexpr int Tn = 2048;
constexpr int Hn = 1024;
constexpr int Gn = 4 * Hn;   // 4096
constexpr int Mn = Bn * Tn;  // 16384 rows

#define AS1 __attribute__((address_space(1)))
#define AS3 __attribute__((address_space(3)))

#if defined(__has_builtin)
#if __has_builtin(__builtin_amdgcn_global_load_async_to_lds_b128)
#define HAVE_ASYNC_LDS 1
#endif
#endif
#ifndef HAVE_ASYNC_LDS
#define HAVE_ASYNC_LDS 0
#endif

__device__ __forceinline__ void wait_async0() {
#if defined(__has_builtin) && __has_builtin(__builtin_amdgcn_s_wait_asynccnt)
  __builtin_amdgcn_s_wait_asynccnt(0);
#else
  asm volatile("s_wait_asynccnt 0" ::: "memory");
#endif
}

// fp32 -> bf16 round-to-nearest-even
__device__ __forceinline__ __bf16 f2bf(float f) {
  unsigned u = __builtin_bit_cast(unsigned, f);
  unsigned r = (u + 0x7FFFu + ((u >> 16) & 1u)) >> 16;
  unsigned short s = (unsigned short)r;
  return __builtin_bit_cast(__bf16, s);
}

__device__ __forceinline__ float sigf(float x) { return 1.0f / (1.0f + __expf(-x)); }

// A fragment: 16x32 bf16, row-major source [*, ld], rows r0..r0+15, K k0..k0+31.
// Lanes 0-15 -> row=lane, K {k0..k0+7} in v0-3, {k0+16..k0+23} in v4-7;
// lanes 16-31 -> K {k0+8..k0+15} and {k0+24..k0+31}.
__device__ __forceinline__ v16bf load_a_frag(const __bf16* base, int ld, int r0, int k0, int lane) {
  const __bf16* p = base + (size_t)(r0 + (lane & 15)) * ld + (k0 + ((lane >> 4) << 3));
  v8bf lo = *(const v8bf*)p;
  v8bf hi = *(const v8bf*)(p + 16);
  v16bf f;
#pragma unroll
  for (int i = 0; i < 8; ++i) { f[i] = lo[i]; f[i + 8] = hi[i]; }
  return f;
}

// B fragment: 32x16 bf16 (K x N), source W stored [N, K] row-major.
// Lane n (0-15) holds column n0+n, K {k0..k0+7} and {k0+16..k0+23}; lanes 16-31 hold K {k0+8..k0+15} and {k0+24..k0+31}.
__device__ __forceinline__ v16bf load_b_frag(const __bf16* base, int ld, int n0, int k0, int lane) {
  const __bf16* p = base + (size_t)(n0 + (lane & 15)) * ld + (k0 + ((lane >> 4) << 3));
  v8bf lo = *(const v8bf*)p;
  v8bf hi = *(const v8bf*)(p + 16);
  v16bf f;
#pragma unroll
  for (int i = 0; i < 8; ++i) { f[i] = lo[i]; f[i + 8] = hi[i]; }
  return f;
}

__device__ __forceinline__ v8f wmma_bf16(v16bf a, v16bf b, v8f c) {
  return __builtin_amdgcn_wmma_f32_16x16x32_bf16(false, a, false, b, (short)0, c, false, false);
}

// ---------------------------------------------------------------------------
// Tile staging: copy a 128-row x 32-col bf16 tile (rows row0.., cols k0..k0+31
// of a [*, ld] row-major matrix) into LDS laid out [128][32] contiguous.
// 256 threads: thread -> (row = tid>>1, half = tid&1), two 16B chunks each.
// ---------------------------------------------------------------------------
#if HAVE_ASYNC_LDS
__device__ __forceinline__ void stage_tile_async(__bf16* dst, const __bf16* src,
                                                 int ld, int row0, int k0, int tid) {
  int row = tid >> 1;
  int half = tid & 1;
  const __bf16* s = src + (size_t)(row0 + row) * ld + k0 + half * 8;
  __bf16* d = dst + row * 32 + half * 8;
  AS1 v4i* g = (AS1 v4i*)s;
  AS3 v4i* l = (AS3 v4i*)d;
  __builtin_amdgcn_global_load_async_to_lds_b128(g, l, 0, 0);
  __builtin_amdgcn_global_load_async_to_lds_b128(g, l, 32, 0);  // imm offset -> both addrs
}
#endif

__device__ __forceinline__ void stage_tile_sync(__bf16* dst, const __bf16* src,
                                                int ld, int row0, int k0, int tid) {
  int row = tid >> 1;
  int half = tid & 1;
  const __bf16* s = src + (size_t)(row0 + row) * ld + k0 + half * 8;
  __bf16* d = dst + row * 32 + half * 8;
  *(v8bf*)d = *(const v8bf*)s;
  *(v8bf*)(d + 16) = *(const v8bf*)(s + 16);
}

// ---------------------------------------------------------------------------
// fp32 -> bf16 weight conversion (vectorized x4)
// ---------------------------------------------------------------------------
__global__ __launch_bounds__(256) void cvt4_kernel(const float* __restrict__ in,
                                                   __bf16* __restrict__ out, int n4) {
  int i = blockIdx.x * 256 + threadIdx.x;
  if (i < n4) {
    float4 v = ((const float4*)in)[i];
    v4bf o;
    o[0] = f2bf(v.x); o[1] = f2bf(v.y); o[2] = f2bf(v.z); o[3] = f2bf(v.w);
    ((v4bf*)out)[i] = o;
  }
}

__global__ __launch_bounds__(256) void bias_sum_kernel(const float* __restrict__ a,
                                                       const float* __restrict__ b,
                                                       float* __restrict__ o, int n) {
  int i = blockIdx.x * 256 + threadIdx.x;
  if (i < n) o[i] = a[i] + b[i];
}

// ---------------------------------------------------------------------------
// LayerNorm over H=1024, one row per block (256 threads, 4 elems/thread),
// output bf16 for WMMA consumption.
// ---------------------------------------------------------------------------
__global__ __launch_bounds__(256) void ln_bf16_kernel(const float* __restrict__ x,
                                                      const float* __restrict__ gamma,
                                                      const float* __restrict__ beta,
                                                      __bf16* __restrict__ out) {
  const int row = blockIdx.x;
  const int tid = threadIdx.x;
  const int lane = tid & 31, wave = tid >> 5;
  const float* xr = x + (size_t)row * Hn;
  float4 v = ((const float4*)xr)[tid];

  __shared__ float red[8];
  __shared__ float stat;

  float s = v.x + v.y + v.z + v.w;
#pragma unroll
  for (int o = 16; o > 0; o >>= 1) s += __shfl_down(s, o, 32);
  if (lane == 0) red[wave] = s;
  __syncthreads();
  if (tid == 0) {
    float t = 0.f;
#pragma unroll
    for (int i = 0; i < 8; ++i) t += red[i];
    stat = t * (1.0f / Hn);
  }
  __syncthreads();
  float m = stat;

  float dx = v.x - m, dy = v.y - m, dz = v.z - m, dw = v.w - m;
  float q = dx * dx + dy * dy + dz * dz + dw * dw;
#pragma unroll
  for (int o = 16; o > 0; o >>= 1) q += __shfl_down(q, o, 32);
  if (lane == 0) red[wave] = q;
  __syncthreads();
  if (tid == 0) {
    float t = 0.f;
#pragma unroll
    for (int i = 0; i < 8; ++i) t += red[i];
    stat = rsqrtf(t * (1.0f / Hn) + 1e-5f);
  }
  __syncthreads();
  float rs = stat;

  float4 g = ((const float4*)gamma)[tid];
  float4 b = ((const float4*)beta)[tid];
  v4bf o4;
  o4[0] = f2bf(dx * rs * g.x + b.x);
  o4[1] = f2bf(dy * rs * g.y + b.y);
  o4[2] = f2bf(dz * rs * g.z + b.z);
  o4[3] = f2bf(dw * rs * g.w + b.w);
  ((v4bf*)out)[(size_t)row * (Hn / 4) + tid] = o4;
}

// ---------------------------------------------------------------------------
// Generic TN GEMM: C[M,N] = A[M,K] (bf16) x B[N,K]^T (bf16), both row-major.
// Block = 256 threads = 8 waves arranged 4(M) x 2(N); wave tile 32x64;
// block tile 128x128; K-step 32. A/B tiles double-buffered in LDS, staged
// with async global->LDS copies; fragments read back with ds_load_b128.
// EPI 0: Cf = acc + bias            (fp32 out)
// EPI 1: Cb = bf16(relu(acc+bias))  (bf16 out)
// EPI 2: Cf = resid + acc + bias    (fp32 out)
// ---------------------------------------------------------------------------
template <int EPI>
__global__ __launch_bounds__(256) void gemm_tn_kernel(const __bf16* __restrict__ A,
                                                      const __bf16* __restrict__ Bm,
                                                      const float* __restrict__ bias,
                                                      const float* __restrict__ resid,
                                                      float* __restrict__ Cf,
                                                      __bf16* __restrict__ Cb,
                                                      int M, int N, int K) {
  __shared__ __bf16 smA[2][128 * 32];  // 8 KB each
  __shared__ __bf16 smB[2][128 * 32];
  const int tid = threadIdx.x;
  const int lane = tid & 31;
  const int wave = tid >> 5;
  const int wy = wave & 3, wx = wave >> 2;
  const int bm = blockIdx.y * 128;
  const int bn = blockIdx.x * 128;

  v8f acc[2][4];
#pragma unroll
  for (int i = 0; i < 2; ++i)
#pragma unroll
    for (int j = 0; j < 4; ++j)
#pragma unroll
      for (int r = 0; r < 8; ++r) acc[i][j][r] = 0.0f;

  // Prologue: stage k-slice 0 into buffer 0.
#if HAVE_ASYNC_LDS
  stage_tile_async(smA[0], A, K, bm, 0, tid);
  stage_tile_async(smB[0], Bm, K, bn, 0, tid);
  wait_async0();
#else
  stage_tile_sync(smA[0], A, K, bm, 0, tid);
  stage_tile_sync(smB[0], Bm, K, bn, 0, tid);
#endif
  __syncthreads();

  const int nsteps = K >> 5;
  for (int s = 0; s < nsteps; ++s) {
    const int cur = s & 1;
    const int nxt = cur ^ 1;
    const bool more = (s + 1) < nsteps;

#if HAVE_ASYNC_LDS
    if (more) {  // overlap next-slice copies with this slice's WMMAs
      stage_tile_async(smA[nxt], A, K, bm, (s + 1) * 32, tid);
      stage_tile_async(smB[nxt], Bm, K, bn, (s + 1) * 32, tid);
    }
#endif

    v16bf a0 = load_a_frag(smA[cur], 32, wy * 32, 0, lane);
    v16bf a1 = load_a_frag(smA[cur], 32, wy * 32 + 16, 0, lane);
    v16bf b0 = load_b_frag(smB[cur], 32, wx * 64 + 0,  0, lane);
    v16bf b1 = load_b_frag(smB[cur], 32, wx * 64 + 16, 0, lane);
    v16bf b2 = load_b_frag(smB[cur], 32, wx * 64 + 32, 0, lane);
    v16bf b3 = load_b_frag(smB[cur], 32, wx * 64 + 48, 0, lane);
    acc[0][0] = wmma_bf16(a0, b0, acc[0][0]);
    acc[0][1] = wmma_bf16(a0, b1, acc[0][1]);
    acc[0][2] = wmma_bf16(a0, b2, acc[0][2]);
    acc[0][3] = wmma_bf16(a0, b3, acc[0][3]);
    acc[1][0] = wmma_bf16(a1, b0, acc[1][0]);
    acc[1][1] = wmma_bf16(a1, b1, acc[1][1]);
    acc[1][2] = wmma_bf16(a1, b2, acc[1][2]);
    acc[1][3] = wmma_bf16(a1, b3, acc[1][3]);

    if (more) {
#if HAVE_ASYNC_LDS
      wait_async0();
#else
      __syncthreads();
      stage_tile_sync(smA[nxt], A, K, bm, (s + 1) * 32, tid);
      stage_tile_sync(smB[nxt], Bm, K, bn, (s + 1) * 32, tid);
#endif
      __syncthreads();
    }
  }

  const int nl = lane & 15;
  const int mh = (lane >> 4) << 3;
#pragma unroll
  for (int i = 0; i < 2; ++i) {
#pragma unroll
    for (int j = 0; j < 4; ++j) {
      int n = bn + wx * 64 + j * 16 + nl;
      float bs = bias[n];
#pragma unroll
      for (int r = 0; r < 8; ++r) {
        int m = bm + wy * 32 + i * 16 + mh + r;
        size_t idx = (size_t)m * N + n;
        float val = acc[i][j][r] + bs;
        if (EPI == 0) {
          Cf[idx] = val;
        } else if (EPI == 1) {
          Cb[idx] = f2bf(val > 0.0f ? val : 0.0f);
        } else {
          Cf[idx] = resid[idx] + val;
        }
      }
    }
  }
}

// ---------------------------------------------------------------------------
// LSTM state init: zero h double-buffers (16x1024 bf16 each; rows 8..15 stay
// zero forever -> benign WMMA padding) and c (8x1024 fp32).
// ---------------------------------------------------------------------------
__global__ __launch_bounds__(256) void lstm_init_kernel(__bf16* hA, __bf16* hB, float* c) {
  int i = blockIdx.x * 256 + threadIdx.x;
  if (i < 16 * Hn) { hA[i] = f2bf(0.0f); hB[i] = f2bf(0.0f); }
  if (i < Bn * Hn) c[i] = 0.0f;
}

// ---------------------------------------------------------------------------
// One fused LSTM timestep. 64 blocks; block owns 16 h-columns across all 4
// gates. The 64x1024 W_hh sub-block (128 KB) is bulk-staged into LDS with
// async copies (no destination VGPRs -> no regalloc serialization; one
// s_wait_asynccnt), padded to a 1032-element pitch so b128 fragment reads are
// LDS-bank-conflict-free (2064 B = 516 dwords = 4 banks/row; 16 lanes x 4
// banks = all 64 banks). WMMAs then consume B from LDS at ds latency.
// 8 waves split K (128 each); partials reduced through LDS (aliasing smW),
// then 128 threads apply the nonlinearities and update c, h, x2 = x + h.
// ---------------------------------------------------------------------------
constexpr int WPITCH = 1032;  // Hn + 8 element pad

__global__ __launch_bounds__(256) void lstm_step_kernel(const __bf16* __restrict__ Whh,
                                                        const float* __restrict__ xg,
                                                        const __bf16* __restrict__ hprev,
                                                        __bf16* __restrict__ hnext,
                                                        float* __restrict__ c,
                                                        const float* __restrict__ x,
                                                        float* __restrict__ x2, int t) {
  __shared__ __bf16 smW[64 * WPITCH];  // ~129 KB
  const int tid = threadIdx.x;
  const int lane = tid & 31;
  const int wave = tid >> 5;
  const int hc0 = blockIdx.x * 16;

  // Preload A fragments (h_prev, L2-hot) while staging runs.
  const int kbase = wave * 128;
  v16bf a[4];
#pragma unroll
  for (int kk = 0; kk < 4; ++kk) a[kk] = load_a_frag(hprev, Hn, 0, kbase + kk * 32, lane);

  // Bulk-stage W_hh rows {g*Hn + hc0 + rc} (64 rows x 1024 cols) into LDS.
  // 8192 16B chunks / 256 threads = 32 per thread.
#pragma unroll 4
  for (int i = 0; i < 32; ++i) {
    int cidx = tid + i * 256;
    int row = cidx >> 7;        // 0..63  (g*16 + rc)
    int col8 = cidx & 127;      // 16B chunk within row
    int g = row >> 4, rc = row & 15;
    const __bf16* s = Whh + (size_t)(g * Hn + hc0 + rc) * Hn + col8 * 8;
    __bf16* d = smW + row * WPITCH + col8 * 8;
#if HAVE_ASYNC_LDS
    __builtin_amdgcn_global_load_async_to_lds_b128((AS1 v4i*)s, (AS3 v4i*)d, 0, 0);
#else
    *(v8bf*)d = *(const v8bf*)s;
#endif
  }
#if HAVE_ASYNC_LDS
  wait_async0();
#endif
  __syncthreads();

  v8f acc[4];
#pragma unroll
  for (int g = 0; g < 4; ++g)
#pragma unroll
    for (int r = 0; r < 8; ++r) acc[g][r] = 0.0f;

#pragma unroll
  for (int kk = 0; kk < 4; ++kk) {
    const int k = kbase + kk * 32;
#pragma unroll
    for (int g = 0; g < 4; ++g) {
      v16bf b = load_b_frag(smW, WPITCH, g * 16, k, lane);
      acc[g] = wmma_bf16(a[kk], b, acc[g]);
    }
  }

  // Cross-wave reduction: reuse smW as the fp32 partial buffer.
  __syncthreads();  // all waves done reading smW
  float* red = (float*)smW;  // needs 8*4*256 floats = 32 KB
  const int nl = lane & 15;
  const int mh = (lane >> 4) << 3;
#pragma unroll
  for (int g = 0; g < 4; ++g)
#pragma unroll
    for (int r = 0; r < 8; ++r)
      red[(wave * 4 + g) * 256 + (mh + r) * 16 + nl] = acc[g][r];
  __syncthreads();

  if (tid < 128) {
    int b = tid >> 4, col = tid & 15;
    int hc = hc0 + col;
    float gate[4];
#pragma unroll
    for (int g = 0; g < 4; ++g) {
      float s = 0.0f;
#pragma unroll
      for (int w = 0; w < 8; ++w) s += red[(w * 4 + g) * 256 + b * 16 + col];
      gate[g] = s + xg[(size_t)(b * Tn + t) * Gn + g * Hn + hc];
    }
    float i_ = sigf(gate[0]);
    float f_ = sigf(gate[1]);
    float g_ = tanhf(gate[2]);
    float o_ = sigf(gate[3]);
    size_t ci = (size_t)b * Hn + hc;
    float cn = f_ * c[ci] + i_ * g_;
    c[ci] = cn;
    float h = o_ * tanhf(cn);
    hnext[ci] = f2bf(h);
    size_t xi = (size_t)(b * Tn + t) * Hn + hc;
    x2[xi] = x[xi] + h;
  }
}

// ---------------------------------------------------------------------------
// Host-side orchestration
// ---------------------------------------------------------------------------
extern "C" void kernel_launch(void* const* d_in, const int* in_sizes, int n_in,
                              void* d_out, int out_size, void* d_ws, size_t ws_size,
                              hipStream_t stream) {
  const float* x    = (const float*)d_in[0];
  const float* l1g  = (const float*)d_in[1];
  const float* l1b  = (const float*)d_in[2];
  const float* Wih  = (const float*)d_in[3];
  const float* Whh  = (const float*)d_in[4];
  const float* bih  = (const float*)d_in[5];
  const float* bhh  = (const float*)d_in[6];
  const float* l2g  = (const float*)d_in[7];
  const float* l2b  = (const float*)d_in[8];
  const float* W1   = (const float*)d_in[9];
  const float* b1   = (const float*)d_in[10];
  const float* W2   = (const float*)d_in[11];
  const float* b2   = (const float*)d_in[12];
  float* out = (float*)d_out;

  // Workspace carve-up
  char* p = (char*)d_ws;
  auto take = [&](size_t bytes) { char* q = p; p += (bytes + 255) & ~(size_t)255; return q; };
  const size_t Wsz = (size_t)Gn * Hn;          // 4M elems per weight
  __bf16* Wih_bf = (__bf16*)take(Wsz * 2);
  __bf16* Whh_bf = (__bf16*)take(Wsz * 2);
  __bf16* W1_bf  = (__bf16*)take(Wsz * 2);
  __bf16* W2_bf  = (__bf16*)take(Wsz * 2);
  __bf16* xn     = (__bf16*)take((size_t)Mn * Hn * 2);   // LN1(x), bf16
  __bf16* y_bf   = (__bf16*)take((size_t)Mn * Hn * 2);   // LN2(x2), bf16
  __bf16* ff_bf  = (__bf16*)take((size_t)Mn * Gn * 2);   // relu FFN hidden, bf16
  float*  xg     = (float*) take((size_t)Mn * Gn * 4);   // input gate pre-acts, fp32
  float*  x2     = (float*) take((size_t)Mn * Hn * 4);   // x + h, fp32
  __bf16* hA     = (__bf16*)take((size_t)16 * Hn * 2);
  __bf16* hB     = (__bf16*)take((size_t)16 * Hn * 2);
  float*  cst    = (float*) take((size_t)Bn * Hn * 4);
  float*  bsum   = (float*) take((size_t)Gn * 4);

  // 1) weights fp32 -> bf16, combined LSTM bias
  cvt4_kernel<<<4096, 256, 0, stream>>>(Wih, Wih_bf, (int)(Wsz / 4));
  cvt4_kernel<<<4096, 256, 0, stream>>>(Whh, Whh_bf, (int)(Wsz / 4));
  cvt4_kernel<<<4096, 256, 0, stream>>>(W1,  W1_bf,  (int)(Wsz / 4));
  cvt4_kernel<<<4096, 256, 0, stream>>>(W2,  W2_bf,  (int)(Wsz / 4));
  bias_sum_kernel<<<16, 256, 0, stream>>>(bih, bhh, bsum, Gn);

  // 2) LN1: xn = LN(x)
  ln_bf16_kernel<<<Mn, 256, 0, stream>>>(x, l1g, l1b, xn);

  // 3) xg = xn @ W_ih^T + (b_ih + b_hh)   [16384 x 4096]
  gemm_tn_kernel<0><<<dim3(Gn / 128, Mn / 128), 256, 0, stream>>>(
      xn, Wih_bf, bsum, nullptr, xg, nullptr, Mn, Gn, Hn);

  // 4) sequential LSTM: 2048 fused timestep kernels
  lstm_init_kernel<<<64, 256, 0, stream>>>(hA, hB, cst);
  for (int t = 0; t < Tn; ++t) {
    const __bf16* hp = (t & 1) ? hB : hA;
    __bf16* hn = (t & 1) ? hA : hB;
    lstm_step_kernel<<<Hn / 16, 256, 0, stream>>>(Whh_bf, xg, hp, hn, cst, x, x2, t);
  }

  // 5) LN2: y = LN(x2)
  ln_bf16_kernel<<<Mn, 256, 0, stream>>>(x2, l2g, l2b, y_bf);

  // 6) ff = relu(y @ W1^T + b1)           [16384 x 4096], bf16
  gemm_tn_kernel<1><<<dim3(Gn / 128, Mn / 128), 256, 0, stream>>>(
      y_bf, W1_bf, b1, nullptr, nullptr, ff_bf, Mn, Gn, Hn);

  // 7) out = x2 + ff @ W2^T + b2          [16384 x 1024], fp32
  gemm_tn_kernel<2><<<dim3(Hn / 128, Mn / 128), 256, 0, stream>>>(
      ff_bf, W2_bf, b2, x2, out, nullptr, Mn, Hn, Gn);
}
